// smg_activation_function_66451734004359
// MI455X (gfx1250) — matrix operand order repaired and checked by reading, not compiled, not run on hardware
//
#include <hip/hip_runtime.h>

typedef __attribute__((ext_vector_type(16))) _Float16 v16h;
typedef __attribute__((ext_vector_type(8)))  float    v8f;

#define HID 90
#define HP  96          // hidden padded to 96 (3 K-steps of 32, 6 N-tiles of 16)
#define KT  3
#define NT  6
#define MT  2           // M-tiles (of 16 rows) per wave
#define LSTRIDE 104     // LDS row stride in halfs (padded for bank spread)
#define ROWS_PER_WAVE (16 * MT)
#define WAVES_PER_BLOCK 8
#define ROWS_PER_BLOCK (ROWS_PER_WAVE * WAVES_PER_BLOCK)   // 256

// ---------------------------------------------------------------------------
// Prep: small constants.  c0[j] = b0[j] + sum_i cv[i]*w0[i+1][j];  a0 = w0 row 0
// ---------------------------------------------------------------------------
__global__ void prep_small(const float* __restrict__ cv, const float* __restrict__ w0,
                           const float* __restrict__ b0, const float* __restrict__ b1,
                           const float* __restrict__ b2, const float* __restrict__ b3,
                           const float* __restrict__ b4, const float* __restrict__ w5,
                           const float* __restrict__ b5,
                           float* __restrict__ c0, float* __restrict__ biases,
                           float* __restrict__ w5f, float* __restrict__ b5f,
                           _Float16* __restrict__ a0) {
    int j = threadIdx.x;
    if (j < HP) {
        float c = 0.f, a = 0.f, w5v = 0.f;
        if (j < HID) {
            c = b0[j];
            #pragma unroll
            for (int i = 0; i < 6; ++i) c += cv[i] * w0[(i + 1) * HID + j];
            a   = w0[j];
            w5v = w5[j];
        }
        c0[j]  = c;
        a0[j]  = (_Float16)a;
        w5f[j] = w5v;
        const float* bs[4] = {b1, b2, b3, b4};
        #pragma unroll
        for (int l = 0; l < 4; ++l) biases[l * HP + j] = (j < HID) ? bs[l][j] : 0.f;
    }
    if (threadIdx.x == 0) b5f[0] = b5[0];
}

// ---------------------------------------------------------------------------
// Prep: swizzle middle-layer weights into WMMA B-fragment order (f16, padded).
// B (32x16, 16-bit): lanes 0-15 hold K=0..7,16..23 (2 per VGPR), lanes 16-31 K=8..15,24..31.
// => per lane a fragment is 16 K (two runs of 8) at fixed n: one v16h load.
// Layout: wfrag[layer][kt][nt][lane][16]
// ---------------------------------------------------------------------------
__global__ void prep_wfrag(const float* __restrict__ w1, const float* __restrict__ w2,
                           const float* __restrict__ w3, const float* __restrict__ w4,
                           _Float16* __restrict__ wfrag) {
    int idx = blockIdx.x * blockDim.x + threadIdx.x;
    if (idx >= 4 * KT * NT * 32 * 16) return;
    int val = idx & 15;  int t = idx >> 4;
    int ln  = t & 31;    t >>= 5;
    int nt  = t % NT;    t /= NT;
    int kt  = t % KT;    int l = t / KT;
    int k = kt * 32 + (ln >> 4) * 8 + (val >> 3) * 16 + (val & 7);
    int n = nt * 16 + (ln & 15);
    const float* w = (l == 0) ? w1 : (l == 1) ? w2 : (l == 2) ? w3 : w4;
    float v = (k < HID && n < HID) ? w[k * HID + n] : 0.f;
    wfrag[idx] = (_Float16)v;
}

// ---------------------------------------------------------------------------
// Main: 32 rows (2 M-tiles) per wave, 8 waves per block, in-place LDS buffer.
// No block barriers: each wave's buffer is private and wave LDS ops are
// processed in order, and all A-fragments are register-resident before the
// epilogue overwrites the buffer.
// ---------------------------------------------------------------------------
__global__ __launch_bounds__(256) void mlp_main(
    const float* __restrict__ x,
    const _Float16* __restrict__ a0, const float* __restrict__ c0,
    const _Float16* __restrict__ wfrag, const float* __restrict__ biases,
    const float* __restrict__ w5f, const float* __restrict__ b5f,
    float* __restrict__ out) {
    __shared__ _Float16 hbuf[WAVES_PER_BLOCK][ROWS_PER_WAVE * LSTRIDE];

    const int wave = threadIdx.x >> 5;
    const int lane = threadIdx.x & 31;
    const int half16 = lane >> 4;      // 0 or 1
    const int mrow   = lane & 15;
    const long rowBase = (long)blockIdx.x * ROWS_PER_BLOCK + (long)wave * ROWS_PER_WAVE;

    _Float16* cur = hbuf[wave];

    // ---- Layer 0: h0[m][j] = relu(x[m]*a0[j] + c0[j]) ----
    float a0r[3], c0r[3];
    #pragma unroll
    for (int t = 0; t < 3; ++t) {
        a0r[t] = (float)a0[lane + t * 32];
        c0r[t] = c0[lane + t * 32];
    }
    #pragma unroll 4
    for (int m = 0; m < ROWS_PER_WAVE; ++m) {
        float xv = x[rowBase + m];
        #pragma unroll
        for (int t = 0; t < 3; ++t) {
            float v = fmaf(xv, a0r[t], c0r[t]);
            v = v > 0.f ? v : 0.f;
            cur[m * LSTRIDE + lane + t * 32] = (_Float16)v;
        }
    }

    // ---- Layers 1..4: (2x) 16x96 @ 96x96 via v_wmma_f32_16x16x32_f16 ----
    union AFrag { v16h v; unsigned int u[8]; };

    for (int layer = 0; layer < 4; ++layer) {
        // Load ALL A fragments into registers (documented 16-bit A layout):
        // lane M = lane%16; VGPR v holds K pair k0,k0+1 with
        // k0 = (v&3)*2 + (v>>2)*16 + (lane>>4)*8   (+ 32*kt)
        AFrag afr[MT][KT];
        #pragma unroll
        for (int mt = 0; mt < MT; ++mt) {
            #pragma unroll
            for (int kt = 0; kt < KT; ++kt) {
                #pragma unroll
                for (int v = 0; v < 8; ++v) {
                    int k0 = kt * 32 + half16 * 8 + (v >> 2) * 16 + (v & 3) * 2;
                    afr[mt][kt].u[v] =
                        *(const unsigned int*)(cur + (mt * 16 + mrow) * LSTRIDE + k0);
                }
            }
        }

        v8f acc[MT][NT];
        #pragma unroll
        for (int mt = 0; mt < MT; ++mt)
            #pragma unroll
            for (int nt = 0; nt < NT; ++nt) acc[mt][nt] = (v8f){};

        const _Float16* wl = wfrag + (long)layer * (KT * NT * 32 * 16);
        #pragma unroll
        for (int kt = 0; kt < KT; ++kt) {
            #pragma unroll
            for (int nt = 0; nt < NT; ++nt) {
                v16h bfr = *(const v16h*)(wl + ((kt * NT + nt) * 32 + lane) * 16);
                #pragma unroll
                for (int mt = 0; mt < MT; ++mt) {
                    acc[mt][nt] = __builtin_amdgcn_wmma_f32_16x16x32_f16(
                        false, afr[mt][kt].v, false, bfr, (short)0, acc[mt][nt],
                        false, false);
                }
            }
        }

        // Bias + ReLU + store next activations (f16) in place.
        // C/D layout: VGPR v -> M = v + 8*(lane>>4); N = nt*16 + lane%16
        #pragma unroll
        for (int mt = 0; mt < MT; ++mt) {
            #pragma unroll
            for (int nt = 0; nt < NT; ++nt) {
                int n = nt * 16 + mrow;
                float bn = biases[layer * HP + n];
                #pragma unroll
                for (int v = 0; v < 8; ++v) {
                    float val = acc[mt][nt][v] + bn;
                    val = val > 0.f ? val : 0.f;
                    int m = mt * 16 + v + half16 * 8;
                    cur[m * LSTRIDE + n] = (_Float16)val;
                }
            }
        }
    }

    // ---- Layer 5: out[m] = h[m] . w5 + b5 (split columns across lane halves) ----
    float w5r[48];
    #pragma unroll 8
    for (int j = 0; j < 48; ++j) w5r[j] = w5f[half16 * 48 + j];
    float s[MT];
    #pragma unroll
    for (int mt = 0; mt < MT; ++mt) {
        float acc5 = 0.f;
        int rbase = (mt * 16 + mrow) * LSTRIDE + half16 * 48;
        #pragma unroll 8
        for (int j = 0; j < 48; ++j)
            acc5 = fmaf((float)cur[rbase + j], w5r[j], acc5);
        s[mt] = acc5;
    }
    #pragma unroll
    for (int mt = 0; mt < MT; ++mt) s[mt] += __shfl_xor(s[mt], 16, 32);
    if (half16 == 0) {
        #pragma unroll
        for (int mt = 0; mt < MT; ++mt)
            out[rowBase + mt * 16 + mrow] = s[mt] + b5f[0];
    }
}

// ---------------------------------------------------------------------------
// Scalar tail (handles rows beyond the last full 256-row block; normally 0).
// ---------------------------------------------------------------------------
__global__ void mlp_tail(const float* x, const float* cv,
                         const float* w0, const float* b0, const float* w1, const float* b1,
                         const float* w2, const float* b2, const float* w3, const float* b3,
                         const float* w4, const float* b4, const float* w5, const float* b5,
                         float* out, long base, int count) {
    int t = blockIdx.x * blockDim.x + threadIdx.x;
    if (t >= count) return;
    long i = base + t;
    float xv = x[i];
    float h[HID], h2[HID];
    for (int j = 0; j < HID; ++j) {
        float s = b0[j] + xv * w0[j];
        for (int c = 0; c < 6; ++c) s += cv[c] * w0[(c + 1) * HID + j];
        h[j] = fmaxf(s, 0.f);
    }
    const float* ws[4] = {w1, w2, w3, w4};
    const float* bs[4] = {b1, b2, b3, b4};
    for (int l = 0; l < 4; ++l) {
        for (int n = 0; n < HID; ++n) {
            float s = bs[l][n];
            for (int k = 0; k < HID; ++k) s += h[k] * ws[l][k * HID + n];
            h2[n] = fmaxf(s, 0.f);
        }
        for (int n = 0; n < HID; ++n) h[n] = h2[n];
    }
    float s = b5[0];
    for (int k = 0; k < HID; ++k) s += h[k] * w5[k];
    out[i] = s;
}

// ---------------------------------------------------------------------------
extern "C" void kernel_launch(void* const* d_in, const int* in_sizes, int n_in,
                              void* d_out, int out_size, void* d_ws, size_t ws_size,
                              hipStream_t stream) {
    const float* x  = (const float*)d_in[0];
    const float* cv = (const float*)d_in[1];
    const float* w0 = (const float*)d_in[2];
    const float* b0 = (const float*)d_in[3];
    const float* w1 = (const float*)d_in[4];
    const float* b1 = (const float*)d_in[5];
    const float* w2 = (const float*)d_in[6];
    const float* b2 = (const float*)d_in[7];
    const float* w3 = (const float*)d_in[8];
    const float* b3 = (const float*)d_in[9];
    const float* w4 = (const float*)d_in[10];
    const float* b4 = (const float*)d_in[11];
    const float* w5 = (const float*)d_in[12];
    const float* b5 = (const float*)d_in[13];
    float* out = (float*)d_out;
    long M = (long)out_size;

    char* ws = (char*)d_ws;
    float*    c0     = (float*)(ws + 0);       // 96 f32
    float*    biases = (float*)(ws + 512);     // 4*96 f32
    float*    w5f    = (float*)(ws + 2048);    // 96 f32
    float*    b5f    = (float*)(ws + 2432);    // 1 f32
    _Float16* a0     = (_Float16*)(ws + 2560); // 96 f16
    _Float16* wfrag  = (_Float16*)(ws + 4096); // 4*3*6*32*16 f16 = 73728 B

    prep_small<<<1, 128, 0, stream>>>(cv, w0, b0, b1, b2, b3, b4, w5, b5,
                                      c0, biases, w5f, b5f, a0);
    prep_wfrag<<<(4 * KT * NT * 32 * 16 + 255) / 256, 256, 0, stream>>>(w1, w2, w3, w4, wfrag);

    long full = M / ROWS_PER_BLOCK;
    if (full > 0)
        mlp_main<<<(int)full, 256, 0, stream>>>(x, a0, c0, wfrag, biases, w5f, b5f, out);
    long rem = M - full * ROWS_PER_BLOCK;
    if (rem > 0)
        mlp_tail<<<(int)((rem + 63) / 64), 64, 0, stream>>>(
            x, cv, w0, b0, w1, b1, w2, b2, w3, b3, w4, b4, w5, b5,
            out, full * ROWS_PER_BLOCK, (int)rem);
}
